// DoubleGRUModel_24180665876591
// MI455X (gfx1250) — matrix operand order repaired and checked by reading, not compiled, not run on hardware
//
#include <hip/hip_runtime.h>
#include <hip/hip_bf16.h>

typedef __attribute__((ext_vector_type(8)))  float  v8f;
typedef __attribute__((ext_vector_type(8)))  __bf16 v8bf;
typedef __attribute__((ext_vector_type(16))) __bf16 v16bf;

#define BATCH 64
#define TLEN  2048
#define IDIM  256
#define HDIM  512
#define GDIM  1536          // 3*H
#define BT    16            // batch rows per workgroup (one WMMA M-tile)
#define THREADS 512         // 16 waves (wave32)
#define WAVES (THREADS/32)
#define NTILE_H (HDIM/16)   // 32 column tiles of H
#define TPW (NTILE_H/WAVES) // 2 tiles per wave

// ---- WMMA fragment helpers (wave32, 16x16x32 bf16) ----------------------

// A-matrix 16x32 bf16 from LDS, row-major [16][K]:
// lanes 0-15: row M=lane,   K = {k0..k0+7, k0+16..k0+23}
// lanes16-31: row M=lane-16,K = {k0+8..k0+15, k0+24..k0+31}
__device__ __forceinline__ v16bf load_a_lds(const __bf16* s, int K, int lane, int k0) {
  int m  = lane & 15;
  int kk = k0 + ((lane & 16) ? 8 : 0);
  const v8bf* p0 = (const v8bf*)(s + m * K + kk);
  const v8bf* p1 = (const v8bf*)(s + m * K + kk + 16);
  v8bf lo = *p0, hi = *p1;
  return __builtin_shufflevector(lo, hi, 0,1,2,3,4,5,6,7,8,9,10,11,12,13,14,15);
}

// B-matrix 32x16 bf16 from global weight W[rows][K] (row-major bf16):
// column N = lane&15 is gate-row (row0 + N); lanes 0-15 carry K={k0..k0+7, k0+16..k0+23},
// lanes 16-31 carry K={k0+8..k0+15, k0+24..k0+31} -> two 16B loads per lane.
__device__ __forceinline__ v16bf load_b_glb(const __bf16* g, int K, int lane, int row0, int k0) {
  int r  = row0 + (lane & 15);
  int kk = k0 + ((lane & 16) ? 8 : 0);
  v8bf lo = *(const v8bf*)(g + (size_t)r * K + kk);
  v8bf hi = *(const v8bf*)(g + (size_t)r * K + kk + 16);
  return __builtin_shufflevector(lo, hi, 0,1,2,3,4,5,6,7,8,9,10,11,12,13,14,15);
}

__device__ __forceinline__ v8f wmma_bf(v16bf a, v16bf b, v8f c) {
  return __builtin_amdgcn_wmma_f32_16x16x32_bf16(false, a, false, b, (short)0, c,
                                                 false, false);
}

__device__ __forceinline__ float sigmoidf_(float x) { return 1.f / (1.f + __expf(-x)); }

// ---- weight fp32 -> bf16 conversion into workspace ----------------------

__global__ void convert_weights(const float* __restrict__ wi1, const float* __restrict__ wh1,
                                const float* __restrict__ wi2, const float* __restrict__ wh2,
                                __bf16* __restrict__ dst) {
  const size_t n1 = (size_t)GDIM * IDIM;   // Wi1
  const size_t n2 = (size_t)GDIM * HDIM;   // Wh1 / Wi2 / Wh2
  const size_t total = n1 + 3 * n2;
  for (size_t i = (size_t)blockIdx.x * blockDim.x + threadIdx.x; i < total;
       i += (size_t)gridDim.x * blockDim.x) {
    float v;
    if (i < n1)               v = wi1[i];
    else if (i < n1 + n2)     v = wh1[i - n1];
    else if (i < n1 + 2 * n2) v = wi2[i - n1 - n2];
    else                      v = wh2[i - n1 - 2 * n2];
    dst[i] = (__bf16)v;
  }
}

// ---- persistent 2-layer GRU: 1 workgroup per 16 batch rows --------------

__global__ __launch_bounds__(THREADS, 1)
void gru2_persistent(const float* __restrict__ x,
                     const int*   __restrict__ lengths,
                     const float* __restrict__ state0,
                     const float* __restrict__ state1,
                     const float* __restrict__ bi1, const float* __restrict__ bh1,
                     const float* __restrict__ bi2, const float* __restrict__ bh2,
                     const float* __restrict__ W1,  const float* __restrict__ b1,
                     const float* __restrict__ W2,  const float* __restrict__ b2,
                     const __bf16* __restrict__ wi1b,
                     const __bf16* __restrict__ wh1b,
                     const __bf16* __restrict__ wi2b,
                     const __bf16* __restrict__ wh2b,
                     float* __restrict__ out) {
  __shared__ __align__(32) __bf16 s_xbf[BT * IDIM];   //  8 KB  x_t as bf16
  __shared__ __align__(32) float  s_h1f[BT * HDIM];   // 32 KB  h1 fp32 master
  __shared__ __align__(32) float  s_h2f[BT * HDIM];   // 32 KB  h2 fp32 master
  __shared__ __align__(32) __bf16 s_h1b[BT * HDIM];   // 16 KB  h1 bf16 (WMMA A)
  __shared__ __align__(32) __bf16 s_h2b[BT * HDIM];   // 16 KB  h2 bf16 (WMMA A)
  __shared__ int s_len[BT];

  const int tid  = threadIdx.x;
  const int lane = tid & 31;
  const int wv   = tid >> 5;
  const int b0   = blockIdx.x * BT;

  // init hidden state from learned init vectors
  for (int i = tid; i < BT * HDIM; i += THREADS) {
    int j = i & (HDIM - 1);
    float h1 = state0[j], h2 = state1[j];
    s_h1f[i] = h1;          s_h2f[i] = h2;
    s_h1b[i] = (__bf16)h1;  s_h2b[i] = (__bf16)h2;
  }
  if (tid < BT) s_len[tid] = lengths[b0 + tid];

  // biases depend only on gate column == lane%16 in the C/D layout:
  // hoist out of the T loop into registers (used as accumulator init).
  float Br1[TPW], Bz1[TPW], Bin1[TPW], Bhn1[TPW];
  float Br2[TPW], Bz2[TPW], Bin2[TPW], Bhn2[TPW];
  for (int s = 0; s < TPW; ++s) {
    int jc = (wv * TPW + s) * 16 + (lane & 15);
    Br1[s]  = bi1[jc] + bh1[jc];
    Bz1[s]  = bi1[HDIM + jc] + bh1[HDIM + jc];
    Bin1[s] = bi1[2 * HDIM + jc];
    Bhn1[s] = bh1[2 * HDIM + jc];
    Br2[s]  = bi2[jc] + bh2[jc];
    Bz2[s]  = bi2[HDIM + jc] + bh2[HDIM + jc];
    Bin2[s] = bi2[2 * HDIM + jc];
    Bhn2[s] = bh2[2 * HDIM + jc];
  }
  __syncthreads();

  for (int t = 0; t < TLEN; ++t) {
    // ---- stage x_t[16,256] -> bf16 LDS, prefetch next step -------------
    for (int i = tid; i < BT * IDIM; i += THREADS) {
      int m = i >> 8;            // /IDIM
      int c = i & (IDIM - 1);
      const float* xp = x + ((size_t)(b0 + m) * TLEN + t) * IDIM + c;
      s_xbf[i] = (__bf16)(*xp);
      if (t + 1 < TLEN) __builtin_prefetch(xp + IDIM, 0, 0);
    }
    __syncthreads();

    // ---- layer 1: gates in registers, per-wave column tiles ------------
    float h1n[TPW][8];
    #pragma unroll 1
    for (int s = 0; s < TPW; ++s) {
      const int j0 = (wv * TPW + s) * 16;
      v8f ar, az, ain, ahn;
      #pragma unroll
      for (int q = 0; q < 8; ++q) { ar[q] = Br1[s]; az[q] = Bz1[s]; ain[q] = Bin1[s]; ahn[q] = Bhn1[s]; }
      #pragma unroll 2
      for (int k0 = 0; k0 < IDIM; k0 += 32) {           // x @ Wi1^T
        v16bf a = load_a_lds(s_xbf, IDIM, lane, k0);
        ar  = wmma_bf(a, load_b_glb(wi1b, IDIM, lane, j0,            k0), ar);
        az  = wmma_bf(a, load_b_glb(wi1b, IDIM, lane, HDIM + j0,     k0), az);
        ain = wmma_bf(a, load_b_glb(wi1b, IDIM, lane, 2 * HDIM + j0, k0), ain);
      }
      #pragma unroll 2
      for (int k0 = 0; k0 < HDIM; k0 += 32) {           // h1 @ Wh1^T
        v16bf a = load_a_lds(s_h1b, HDIM, lane, k0);
        ar  = wmma_bf(a, load_b_glb(wh1b, HDIM, lane, j0,            k0), ar);
        az  = wmma_bf(a, load_b_glb(wh1b, HDIM, lane, HDIM + j0,     k0), az);
        ahn = wmma_bf(a, load_b_glb(wh1b, HDIM, lane, 2 * HDIM + j0, k0), ahn);
      }
      const int jc = j0 + (lane & 15);
      const int mh = (lane & 16) ? 8 : 0;
      #pragma unroll
      for (int q = 0; q < 8; ++q) {
        int   m  = mh + q;
        float r  = sigmoidf_(ar[q]);
        float z  = sigmoidf_(az[q]);
        float n  = tanhf(ain[q] + r * ahn[q]);
        float ho = s_h1f[m * HDIM + jc];
        float hv = (1.f - z) * n + z * ho;
        h1n[s][q] = (t < s_len[m]) ? hv : ho;
      }
    }
    __syncthreads();                                    // all layer-1 reads done
    #pragma unroll 1
    for (int s = 0; s < TPW; ++s) {
      int jc = (wv * TPW + s) * 16 + (lane & 15);
      int mh = (lane & 16) ? 8 : 0;
      #pragma unroll
      for (int q = 0; q < 8; ++q) {
        int m = mh + q;
        s_h1f[m * HDIM + jc] = h1n[s][q];
        s_h1b[m * HDIM + jc] = (__bf16)h1n[s][q];
      }
    }
    __syncthreads();                                    // h1 visible for layer 2

    // ---- layer 2 -------------------------------------------------------
    float h2n[TPW][8];
    #pragma unroll 1
    for (int s = 0; s < TPW; ++s) {
      const int j0 = (wv * TPW + s) * 16;
      v8f ar, az, ain, ahn;
      #pragma unroll
      for (int q = 0; q < 8; ++q) { ar[q] = Br2[s]; az[q] = Bz2[s]; ain[q] = Bin2[s]; ahn[q] = Bhn2[s]; }
      #pragma unroll 2
      for (int k0 = 0; k0 < HDIM; k0 += 32) {           // h1n @ Wi2^T
        v16bf a = load_a_lds(s_h1b, HDIM, lane, k0);
        ar  = wmma_bf(a, load_b_glb(wi2b, HDIM, lane, j0,            k0), ar);
        az  = wmma_bf(a, load_b_glb(wi2b, HDIM, lane, HDIM + j0,     k0), az);
        ain = wmma_bf(a, load_b_glb(wi2b, HDIM, lane, 2 * HDIM + j0, k0), ain);
      }
      #pragma unroll 2
      for (int k0 = 0; k0 < HDIM; k0 += 32) {           // h2 @ Wh2^T
        v16bf a = load_a_lds(s_h2b, HDIM, lane, k0);
        ar  = wmma_bf(a, load_b_glb(wh2b, HDIM, lane, j0,            k0), ar);
        az  = wmma_bf(a, load_b_glb(wh2b, HDIM, lane, HDIM + j0,     k0), az);
        ahn = wmma_bf(a, load_b_glb(wh2b, HDIM, lane, 2 * HDIM + j0, k0), ahn);
      }
      const int jc = j0 + (lane & 15);
      const int mh = (lane & 16) ? 8 : 0;
      #pragma unroll
      for (int q = 0; q < 8; ++q) {
        int   m  = mh + q;
        float r  = sigmoidf_(ar[q]);
        float z  = sigmoidf_(az[q]);
        float n  = tanhf(ain[q] + r * ahn[q]);
        float ho = s_h2f[m * HDIM + jc];
        float hv = (1.f - z) * n + z * ho;
        h2n[s][q] = (t < s_len[m]) ? hv : ho;
      }
    }
    __syncthreads();
    #pragma unroll 1
    for (int s = 0; s < TPW; ++s) {
      int jc = (wv * TPW + s) * 16 + (lane & 15);
      int mh = (lane & 16) ? 8 : 0;
      #pragma unroll
      for (int q = 0; q < 8; ++q) {
        int m = mh + q;
        s_h2f[m * HDIM + jc] = h2n[s][q];
        s_h2b[m * HDIM + jc] = (__bf16)h2n[s][q];
      }
    }
    __syncthreads();
  }

  // ---- head: out = (h2 @ W1^T + b1) @ W2^T + b2 (tiny, fp32) -----------
  // reuse s_h1f's storage as y[16][128]
  for (int i = tid; i < BT * 128; i += THREADS) {
    int m = i >> 7, k = i & 127;
    float acc = b1[k];
    const float* wr = W1 + (size_t)k * HDIM;
    for (int j = 0; j < HDIM; ++j) acc += s_h2f[m * HDIM + j] * wr[j];
    s_h1f[i] = acc;
  }
  __syncthreads();
  if (tid < BT) {
    float acc = b2[0];
    for (int k = 0; k < 128; ++k) acc += s_h1f[tid * 128 + k] * W2[k];
    out[b0 + tid] = acc;
  }
}

// ---- host entry ---------------------------------------------------------

extern "C" void kernel_launch(void* const* d_in, const int* in_sizes, int n_in,
                              void* d_out, int out_size, void* d_ws, size_t ws_size,
                              hipStream_t stream) {
  const float* x      = (const float*)d_in[0];
  const int*   len    = (const int*)  d_in[1];
  const float* state0 = (const float*)d_in[2];
  const float* state1 = (const float*)d_in[3];
  const float* Wi1    = (const float*)d_in[4];
  const float* Wh1    = (const float*)d_in[5];
  const float* bi1    = (const float*)d_in[6];
  const float* bh1    = (const float*)d_in[7];
  const float* Wi2    = (const float*)d_in[8];
  const float* Wh2    = (const float*)d_in[9];
  const float* bi2    = (const float*)d_in[10];
  const float* bh2    = (const float*)d_in[11];
  const float* W1     = (const float*)d_in[12];
  const float* b1     = (const float*)d_in[13];
  const float* W2     = (const float*)d_in[14];
  const float* b2     = (const float*)d_in[15];

  // bf16 weight copies in workspace (~5.25 MB): [Wi1 | Wh1 | Wi2 | Wh2]
  __bf16* wi1b = (__bf16*)d_ws;
  __bf16* wh1b = wi1b + (size_t)GDIM * IDIM;
  __bf16* wi2b = wh1b + (size_t)GDIM * HDIM;
  __bf16* wh2b = wi2b + (size_t)GDIM * HDIM;

  convert_weights<<<1024, 256, 0, stream>>>(Wi1, Wh1, Wi2, Wh2, wi1b);

  gru2_persistent<<<BATCH / BT, THREADS, 0, stream>>>(
      x, len, state0, state1, bi1, bh1, bi2, bh2, W1, b1, W2, b2,
      wi1b, wh1b, wi2b, wh2b, (float*)d_out);
}
